// Attention_72095321030782
// MI455X (gfx1250) — matrix operand order repaired and checked by reading, not compiled, not run on hardware
//
#include <hip/hip_runtime.h>
#include <hip/hip_bf16.h>
#include <stdint.h>

// ---------------------------------------------------------------------------
// Problem constants (from reference): B=16, S=1024, DIM=1024, H=16, HD=64
// ---------------------------------------------------------------------------
#define CB   16
#define CS   1024
#define CDIM 1024
#define CH   16
#define CHD  64
#define BS   (CB * CS)          // 16384 rows
#define N3   (3 * CDIM)         // 3072

typedef _Float16 h16;
typedef __attribute__((ext_vector_type(16))) _Float16 v16h;
typedef __attribute__((ext_vector_type(8)))  _Float16 v8h;
typedef __attribute__((ext_vector_type(8)))  float    v8f;

union V16U { v16h v; v8h h[2]; uint32_t u[8]; };

// WMMA f16 -> f32 accumulate, D = A(16x32) * B(32x16) + C
__device__ __forceinline__ v8f wmma_f16(v16h a, v16h b, v8f c) {
    return __builtin_amdgcn_wmma_f32_16x16x32_f16(false, a, false, b,
                                                  (short)0, c, false, false);
}

// A-fragment (16x32 f16, row-major source with given element stride).
// Lane l<16 holds row m, K = kb..kb+15; lane l+16 gets +16.
// Caller passes the full row index (already includes lane&15 mapping).
__device__ __forceinline__ v16h load_a_frag(const h16* base, int row,
                                            long stride, int kb, int lane) {
    const int khalf = (lane >> 4) * 16;
    const h16* p = base + (long)row * stride + kb + khalf;
    V16U r;
    r.h[0] = *(const v8h*)(p);
    r.h[1] = *(const v8h*)(p + 8);
    return r.v;
}

// ---------------------------------------------------------------------------
// Elementwise casts
// ---------------------------------------------------------------------------
__global__ void cast_f32_to_f16(const float* __restrict__ src,
                                h16* __restrict__ dst, long n) {
    long i = (long)blockIdx.x * blockDim.x + threadIdx.x;
    long stride = (long)gridDim.x * blockDim.x;
    for (; i < n; i += stride) dst[i] = (h16)src[i];
}

// src [K,N] f32 row-major  ->  dst [N,K] f16 row-major (i.e. transpose)
__global__ void transpose_f32_to_f16(const float* __restrict__ src,
                                     h16* __restrict__ dst, int K, int N) {
    long n = (long)N * K;
    long i = (long)blockIdx.x * blockDim.x + threadIdx.x;
    long stride = (long)gridDim.x * blockDim.x;
    for (; i < n; i += stride) {
        int col = (int)(i / K);   // output row = original column
        int k   = (int)(i % K);
        dst[i] = (h16)src[(long)k * N + col];
    }
}

// ---------------------------------------------------------------------------
// Tiled WMMA GEMM:  C[M,N] = A[M,K] * BT[N,K]^T + bias[N]
// One wave computes a 16x64 strip. A,BT f16; C f32 or f16.
// All 5 fragment loads for a k-step are issued as one burst, then 4 WMMAs.
// ---------------------------------------------------------------------------
template <bool HALF_OUT>
__global__ __launch_bounds__(256)
void gemm_wmma(const h16* __restrict__ A, const h16* __restrict__ BT,
               const float* __restrict__ bias, void* __restrict__ Cout,
               int M, int N, int K) {
    const int lane = threadIdx.x & 31;
    const int wave = blockIdx.x * (blockDim.x >> 5) + (threadIdx.x >> 5);
    const int strips = N >> 6;
    const int mt = wave / strips;
    const int nb = (wave - mt * strips) << 6;
    if (mt * 16 >= M) return;             // wave-uniform guard

    v8f acc[4] = {v8f{}, v8f{}, v8f{}, v8f{}};
    const int arow  = mt * 16 + (lane & 15);
    const int bkofs = (lane >> 4) * 16;   // B frag: hi lanes take K+16..K+31
    const h16* brow = BT + (long)(nb + (lane & 15)) * K + bkofs;

    for (int kk = 0; kk < K; kk += 32) {
        // burst all loads first so they clause and overlap
        v16h a = load_a_frag(A, arow, K, kk, lane);
        v16h b[4];
#pragma unroll
        for (int t = 0; t < 4; ++t)
            b[t] = *(const v16h*)(brow + (long)t * 16 * K + kk);
#pragma unroll
        for (int t = 0; t < 4; ++t)
            acc[t] = wmma_f16(a, b[t], acc[t]);
    }

    // C layout: lane<16 -> col=lane, rows 0..7 ; lane>=16 -> col=lane-16, rows 8..15
    const int cq   = lane & 15;
    const int rofs = (lane >> 4) * 8;
#pragma unroll
    for (int t = 0; t < 4; ++t) {
        const int n = nb + t * 16 + cq;
        const float bv = bias ? bias[n] : 0.0f;
#pragma unroll
        for (int r = 0; r < 8; ++r) {
            const long idx = (long)(mt * 16 + rofs + r) * N + n;
            const float v = acc[t][r] + bv;
            if (HALF_OUT) ((h16*)Cout)[idx] = (h16)v;
            else          ((float*)Cout)[idx] = v;
        }
    }
}

// ---------------------------------------------------------------------------
// Per-head RMSNorm + axial 2D RoPE on q/k; v passes through to transposed
// layout. One wave per (b,s,h) row of HD=64 (2 elements per lane).
//   qkv  : [B,S,3,H,HD] f16   (natural [B*S, 3*DIM] GEMM output)
//   qh,kh: [B,H,S,HD]  f16
//   vT   : [B,H,HD,S]  f16
// ---------------------------------------------------------------------------
__global__ __launch_bounds__(256)
void normrope_kernel(const h16* __restrict__ qkv,
                     const float* __restrict__ qw, const float* __restrict__ kw,
                     h16* __restrict__ qh, h16* __restrict__ kh,
                     h16* __restrict__ vT) {
    const int lane = threadIdx.x & 31;
    const int wave = blockIdx.x * (blockDim.x >> 5) + (threadIdx.x >> 5);
    const int h = wave % CH;
    const int s = (wave / CH) % CS;
    const int b = wave / (CH * CS);

    const h16* base = qkv + (long)(b * CS + s) * N3 + h * CHD;
    const int d0 = lane * 2;                     // this lane's two dims

    // --- RoPE angle: f = concat(fh, fw); t = (d<32)? s/32 : s%32 ; j = d&31
    const int fr = s >> 5, fc = s & 31;
    const int j  = d0 & 31;                      // even; d0+1 shares freq & t
    const int t  = (d0 < 32) ? fr : fc;
    const float freq = __powf(10000.0f, -(float)(j & ~1) / 32.0f);
    const float ang  = (float)t * 0.5f * freq;   // t/FT*PT = t*0.5
    float sn, cs;
    __sincosf(ang, &sn, &cs);
    const float sgn = (lane < 16) ? -1.0f : 1.0f;  // rotate_half sign

    const long out_row = ((long)(b * CH + h) * CS + s) * CHD + d0;

#pragma unroll
    for (int comp = 0; comp < 2; ++comp) {       // 0: q, 1: k
        const h16* p = base + comp * CDIM;
        const float x0 = (float)p[d0], x1 = (float)p[d0 + 1];
        float ss = x0 * x0 + x1 * x1;
#pragma unroll
        for (int o = 16; o; o >>= 1) ss += __shfl_xor(ss, o);
        const float rs = rsqrtf(ss * (1.0f / 64.0f) + 1e-6f);
        const float* wv = comp ? kw : qw;
        const float n0 = x0 * rs * wv[d0];
        const float n1 = x1 * rs * wv[d0 + 1];
        const float r0 = __shfl_xor(n0, 16);     // value at d ± 32
        const float r1 = __shfl_xor(n1, 16);
        const float y0 = n0 * cs + sgn * r0 * sn;
        const float y1 = n1 * cs + sgn * r1 * sn;
        h16* outp = (comp ? kh : qh) + out_row;
        outp[0] = (h16)y0;
        outp[1] = (h16)y1;
    }

    // v: transpose into [B,H,HD,S]
    const h16* pv = base + 2 * CDIM;
    h16* vtp = vT + ((long)(b * CH + h) * CHD + d0) * CS + s;
    vtp[0]  = pv[d0];
    vtp[CS] = pv[d0 + 1];
}

// ---------------------------------------------------------------------------
// Flash attention, one wave per 16-query tile of one (b,h).
// Computes S^T = K * Q^T so the softmax row (keys) lives inside a lane,
// then out^T = V^T * P^T. Only lane<->lane^16 exchanges are needed.
// Software-pipelined: next tile's K fragments and this tile's V fragments
// are loaded before the softmax VALU section, which hides their latency.
//   qh,kh: [B,H,S,HD] f16 ; vT: [B,H,HD,S] f16 ; oh: [B,S,DIM] f16
// ---------------------------------------------------------------------------
__global__ __launch_bounds__(256)
void attn_kernel(const h16* __restrict__ qh, const h16* __restrict__ kh,
                 const h16* __restrict__ vT, h16* __restrict__ oh) {
    const int lane = threadIdx.x & 31;
    const int wave = blockIdx.x * (blockDim.x >> 5) + (threadIdx.x >> 5);
    const int qt = wave & 63;                    // S/16 = 64 query tiles
    const int bh = wave >> 6;                    // b*H + h
    const int hh = bh & 15;
    const int bb = bh >> 4;

    const h16* qbase = qh + ((long)bh * CS + qt * 16) * CHD;
    const h16* kbase = kh + (long)bh * CS * CHD;
    const h16* vbase = vT + (long)bh * CHD * CS;

    // Q^T B-fragments (loaded once): lane<16 -> col q=lane, K(=d) 0..15, etc.
    const int qrow = lane & 15;
    const int dsel = (lane >> 4) * 16;
    const v16h bq0 = *(const v16h*)(qbase + qrow * CHD + dsel);        // d 0..31
    const v16h bq1 = *(const v16h*)(qbase + qrow * CHD + 32 + dsel);   // d 32..63

    float m = -__builtin_inff();
    float l = 0.0f;
    v8f o[4] = {v8f{}, v8f{}, v8f{}, v8f{}};     // out^T, 4 d-tiles of 16

    const int krow = lane & 15;
    const int drow = lane & 15;

    // preload K fragments for the first key tile
    v16h k00 = load_a_frag(kbase, krow,      CHD, 0,  lane);
    v16h k01 = load_a_frag(kbase, krow,      CHD, 32, lane);
    v16h k10 = load_a_frag(kbase, 16 + krow, CHD, 0,  lane);
    v16h k11 = load_a_frag(kbase, 16 + krow, CHD, 32, lane);

    for (int kt = 0; kt < CS; kt += 32) {
        // ---- S^T tile (32 keys x 16 queries) from the preloaded K frags ----
        v8f slo = {}, shi = {};
        slo = wmma_f16(k00, bq0, slo);
        slo = wmma_f16(k01, bq1, slo);
        shi = wmma_f16(k10, bq0, shi);
        shi = wmma_f16(k11, bq1, shi);

        // ---- issue this tile's V fragments + next tile's K fragments now ----
        v16h av[4];
#pragma unroll
        for (int t = 0; t < 4; ++t)
            av[t] = load_a_frag(vbase, t * 16 + drow, CS, kt, lane);
        const int kn = kt + 32;
        if (kn < CS) {
            k00 = load_a_frag(kbase, kn + krow,      CHD, 0,  lane);
            k01 = load_a_frag(kbase, kn + krow,      CHD, 32, lane);
            k10 = load_a_frag(kbase, kn + 16 + krow, CHD, 0,  lane);
            k11 = load_a_frag(kbase, kn + 16 + krow, CHD, 32, lane);
            if (kn + 32 < CS) {              // nudge the tile after that into L1
                __builtin_prefetch(kbase + (long)(kn + 32 + krow) * CHD, 0, 1);
                __builtin_prefetch(vbase + (long)drow * CS + kn + 32, 0, 1);
            }
        }

        // ---- streaming softmax (row = fixed query per lane pair) ----
        const float sc = 0.125f;                 // 1/sqrt(64)
        float plo[8], phi[8];
        float tmax = -__builtin_inff();
#pragma unroll
        for (int r = 0; r < 8; ++r) {
            plo[r] = slo[r] * sc;
            phi[r] = shi[r] * sc;
            tmax = fmaxf(tmax, fmaxf(plo[r], phi[r]));
        }
        tmax = fmaxf(tmax, __shfl_xor(tmax, 16));
        const float mn = fmaxf(m, tmax);
        const float alpha = __expf(m - mn);
        m = mn;
        float tsum = 0.0f;
#pragma unroll
        for (int r = 0; r < 8; ++r) {
            plo[r] = __expf(plo[r] - m);
            phi[r] = __expf(phi[r] - m);
            tsum += plo[r] + phi[r];
        }
        tsum += __shfl_xor(tsum, 16);
        l = l * alpha + tsum;
#pragma unroll
        for (int t = 0; t < 4; ++t)
#pragma unroll
            for (int r = 0; r < 8; ++r) o[t][r] *= alpha;

        // ---- build P^T B-fragment (32 keys x 16 queries) ----
        // lane<16 holds k 0..7 (lo) & 16..23 (hi); partner lane+16 has 8..15 & 24..31.
        union { uint32_t u[4]; h16 e[8]; } hlo, hhi;
#pragma unroll
        for (int r = 0; r < 8; ++r) { hlo.e[r] = (h16)plo[r]; hhi.e[r] = (h16)phi[r]; }
        V16U bp;
#pragma unroll
        for (int i = 0; i < 4; ++i) {
            const uint32_t send = (lane < 16) ? hhi.u[i] : hlo.u[i];
            const uint32_t recv = __shfl_xor(send, 16);
            bp.u[i]     = (lane < 16) ? hlo.u[i] : recv;     // K 0..7 / 16..23
            bp.u[4 + i] = (lane < 16) ? recv     : hhi.u[i]; // K 8..15 / 24..31
        }

        // ---- out^T += V^T * P^T (V frags loaded way back, latency hidden) ----
#pragma unroll
        for (int t = 0; t < 4; ++t)
            o[t] = wmma_f16(av[t], bp.v, o[t]);
    }

    // ---- epilogue: scale by 1/l, store packed f16 rows of out ----
    const float linv = 1.0f / l;
    const int q    = lane & 15;
    const int rofs = (lane >> 4) * 8;
    const long orow = ((long)bb * CS + qt * 16 + q) * CDIM + hh * CHD + rofs;
#pragma unroll
    for (int t = 0; t < 4; ++t) {
        union { v8h v; h16 e[8]; } pk;
#pragma unroll
        for (int r = 0; r < 8; ++r) pk.e[r] = (h16)(o[t][r] * linv);
        *(v8h*)(oh + orow + t * 16) = pk.v;
    }
}

// ---------------------------------------------------------------------------
// Host-side launch
// ---------------------------------------------------------------------------
static void* ws_take(void* base, size_t& off, size_t bytes) {
    void* p = (char*)base + off;
    off += (bytes + 255) & ~(size_t)255;
    return p;
}

extern "C" void kernel_launch(void* const* d_in, const int* in_sizes, int n_in,
                              void* d_out, int out_size, void* d_ws, size_t ws_size,
                              hipStream_t stream) {
    const float* x      = (const float*)d_in[0];   // [B,S,DIM]
    const float* w_qkv  = (const float*)d_in[1];   // [DIM,3*DIM]
    const float* b_qkv  = (const float*)d_in[2];   // [3*DIM]
    const float* qnw    = (const float*)d_in[3];   // [HD]
    const float* knw    = (const float*)d_in[4];   // [HD]
    const float* w_out  = (const float*)d_in[5];   // [DIM,DIM]
    const float* b_out  = (const float*)d_in[6];   // [DIM]
    float* out = (float*)d_out;                    // [B,S,DIM] f32

    size_t off = 0;
    h16* xh     = (h16*)ws_take(d_ws, off, (size_t)BS * CDIM * sizeof(h16));
    h16* wqkvT  = (h16*)ws_take(d_ws, off, (size_t)N3 * CDIM * sizeof(h16));
    h16* woutT  = (h16*)ws_take(d_ws, off, (size_t)CDIM * CDIM * sizeof(h16));
    h16* qkvh   = (h16*)ws_take(d_ws, off, (size_t)BS * N3 * sizeof(h16));
    h16* qhbuf  = (h16*)ws_take(d_ws, off, (size_t)BS * CDIM * sizeof(h16));
    h16* khbuf  = (h16*)ws_take(d_ws, off, (size_t)BS * CDIM * sizeof(h16));
    h16* vTbuf  = (h16*)ws_take(d_ws, off, (size_t)BS * CDIM * sizeof(h16));
    h16* attnh  = (h16*)ws_take(d_ws, off, (size_t)BS * CDIM * sizeof(h16));
    (void)ws_size; (void)in_sizes; (void)n_in; (void)out_size;

    const int TPB = 256;

    // 1) casts / transposes to f16
    cast_f32_to_f16<<<4096, TPB, 0, stream>>>(x, xh, (long)BS * CDIM);
    transpose_f32_to_f16<<<4096, TPB, 0, stream>>>(w_qkv, wqkvT, CDIM, N3);
    transpose_f32_to_f16<<<2048, TPB, 0, stream>>>(w_out, woutT, CDIM, CDIM);

    // 2) QKV GEMM: [BS,3*DIM] f16 = xh @ wqkvT^T + b_qkv
    {
        const int waves = (BS / 16) * (N3 / 64);       // 49152
        gemm_wmma<true><<<waves / 8, TPB, 0, stream>>>(xh, wqkvT, b_qkv, qkvh,
                                                       BS, N3, CDIM);
    }

    // 3) RMSNorm + RoPE + V transpose
    {
        const int waves = CB * CS * CH;                // 262144
        normrope_kernel<<<waves / 8, TPB, 0, stream>>>(qkvh, qnw, knw,
                                                       qhbuf, khbuf, vTbuf);
    }

    // 4) Flash attention -> attnh [B,S,DIM] f16
    {
        const int waves = CB * CH * (CS / 16);         // 16384
        attn_kernel<<<waves / 8, TPB, 0, stream>>>(qhbuf, khbuf, vTbuf, attnh);
    }

    // 5) Output projection: out f32 = attnh @ woutT^T + b_out
    {
        const int waves = (BS / 16) * (CDIM / 64);     // 16384
        gemm_wmma<false><<<waves / 8, TPB, 0, stream>>>(attnh, woutT, b_out, out,
                                                        BS, CDIM, CDIM);
    }
}
